// Linear_mvm_4767413699367
// MI455X (gfx1250) — 8-bit matrix operand order repaired and checked by reading, compiled, not run on hardware
//
#include <hip/hip_runtime.h>
#include <stdint.h>

typedef int v8i __attribute__((ext_vector_type(8)));

#define B_SZ   32
#define IN_F   1024
#define OUT_F  1024
#define XBARN  64
#define XR     (IN_F / XBARN)
#define NCOL   (OUT_F * 8)
#define NTILES (NCOL / 16)


__global__ void prep_bits(const float* __restrict__ x, uint32_t* __restrict__ apk) {
  const int b = blockIdx.x >> 4;
  const int r = blockIdx.x & 15;
  __shared__ int xu[XBARN];
  const int t = threadIdx.x;
  if (t < XBARN) {
    float xf = rintf(x[(size_t)b * IN_F + r * XBARN + t] * 4096.0f);
    xf = fminf(fmaxf(xf, -32768.0f), 32767.0f);
    xu[t] = ((int)xf) & 0xFFFF;
  }
  __syncthreads();
  const int lane = t >> 3, v = t & 7;
  const int s = lane & 15, h = lane >> 4;
  const int kb = ((v >> 2) << 5) + ((v & 3) << 2) + (h << 4);
  uint32_t dw = 0;
#pragma unroll
  for (int j = 0; j < 4; ++j)
    dw |= (uint32_t)((xu[kb + j] >> s) & 1) << (8 * j);
  apk[(size_t)blockIdx.x * 256 + lane * 8 + v] = dw;
}

__global__ void prep_slices(const float* __restrict__ w, uint32_t* __restrict__ bpk) {
  const int r     = blockIdx.x >> 9;
  const int ntile = blockIdx.x & 511;
  __shared__ unsigned short wp[2][XBARN], wn[2][XBARN];
  const int t = threadIdx.x;
  if (t < 128) {
    const int oi = t >> 6, d = t & 63;
    const float wv = w[(size_t)(ntile * 2 + oi) * IN_F + r * XBARN + d];
    float fp = rintf(fmaxf(wv, 0.0f) * 4096.0f);
    fp = fminf(fmaxf(fp, 0.0f), 65535.0f);
    float fn = rintf(fabsf(fminf(wv, 0.0f)) * 4096.0f);
    fn = fminf(fmaxf(fn, 0.0f), 65535.0f);
    wp[oi][d] = (unsigned short)fp;
    wn[oi][d] = (unsigned short)fn;
  }
  __syncthreads();
  const int lane = t >> 3, v = t & 7;
  const int n = lane & 15, h = lane >> 4;
  const int oi = n >> 3, sw = n & 7;
  const int shift = 2 * (7 - sw);
  const int kb = ((v >> 2) << 5) + ((v & 3) << 2) + (h << 4);
  uint32_t dwp = 0, dwn = 0;
#pragma unroll
  for (int j = 0; j < 4; ++j) {
    const int d = kb + j;
    dwp |= (uint32_t)((wp[oi][d] >> shift) & 3) << (8 * j);
    dwn |= (uint32_t)((wn[oi][d] >> shift) & 3) << (8 * j);
  }
  const size_t tile = (size_t)r * NTILES + ntile;
  bpk[tile * 256 + lane * 8 + v] = dwp;
  bpk[((size_t)XR * NTILES + tile) * 256 + lane * 8 + v] = dwn;
}

__device__ __forceinline__ int adc_levels(int n) {
  const int t = n * 85;
  return (t + 31 + ((t >> 6) & 1)) >> 6;
}

__device__ __forceinline__ int adc_combine(const v8i d, const bool upper) {
  int s = 0;
#pragma unroll
  for (int e = 0; e < 8; ++e) {
    const int A = adc_levels(d[e]);
    if (upper && e == 7) s -= (A << 7);
    else                 s += (A << e);
  }
  return s;
}

__device__ __forceinline__ long long shflxor64(long long v, int m) {
  int lo = (int)(unsigned long long)v;
  int hi = (int)((unsigned long long)v >> 32);
  lo = __shfl_xor(lo, m, 32);
  hi = __shfl_xor(hi, m, 32);
  return (long long)(((unsigned long long)(unsigned int)hi << 32) | (unsigned int)lo);
}

__device__ __forceinline__ float acm_q(float x) {
  float t = rintf(x * 4096.0f);
  t = fminf(fmaxf(t, -32768.0f), 32767.0f);
  return t;
}

__global__ void __launch_bounds__(256)
mvm_wmma(const uint32_t* __restrict__ apk, const uint32_t* __restrict__ bpk,
         const float* __restrict__ bias, float* __restrict__ out) {
  const int lane  = threadIdx.x & 31;
  const int wave  = threadIdx.x >> 5;
  const int ntile = blockIdx.x >> 2;
  const int b     = ((blockIdx.x & 3) << 3) + wave;
  const bool upper = lane >= 16;
  const int  sh    = 2 * (7 - (lane & 7)) + (upper ? 8 : 0);

  long long accP = 0, accN = 0;
  const v8i cz = {0, 0, 0, 0, 0, 0, 0, 0};

  const uint32_t* ap  = apk + ((size_t)(b * XR) * 32 + lane) * 8;
  const uint32_t* bpP = bpk + ((size_t)ntile * 32 + lane) * 8;
  const uint32_t* bpN = bpP + (size_t)XR * NTILES * 256;

#define LOAD_T(dst_a, dst_p, dst_n, r)                                   \
  dst_a = *(const v8i*)(ap  + (size_t)(r) * 256);                        \
  dst_p = *(const v8i*)(bpP + (size_t)(r) * NTILES * 256);               \
  dst_n = *(const v8i*)(bpN + (size_t)(r) * NTILES * 256)

#define WMMA_IU8(A, B) \
  __builtin_amdgcn_wmma_i32_16x16x64_iu8(false, A, false, B, cz, false, false)

  v8i a0, tp0, tn0, a1, tp1, tn1;
  LOAD_T(a0, tp0, tn0, 0);
  v8i dpA = WMMA_IU8(a0, tp0);
  v8i dnA = WMMA_IU8(a0, tn0);
  LOAD_T(a1, tp1, tn1, 1);
  v8i dpB = WMMA_IU8(a1, tp1);
  v8i dnB = WMMA_IU8(a1, tn1);

  for (int r = 2; r < XR; r += 2) {
    LOAD_T(a0, tp0, tn0, r);
    const int spA = adc_combine(dpA, upper);
    const int snA = adc_combine(dnA, upper);
    dpA = WMMA_IU8(a0, tp0);
    dnA = WMMA_IU8(a0, tn0);
    accP += (long long)spA << sh;
    accN += (long long)snA << sh;

    LOAD_T(a1, tp1, tn1, r + 1);
    const int spB = adc_combine(dpB, upper);
    const int snB = adc_combine(dnB, upper);
    dpB = WMMA_IU8(a1, tp1);
    dnB = WMMA_IU8(a1, tn1);
    accP += (long long)spB << sh;
    accN += (long long)snB << sh;
  }
  accP += (long long)adc_combine(dpA, upper) << sh;
  accN += (long long)adc_combine(dnA, upper) << sh;
  accP += (long long)adc_combine(dpB, upper) << sh;
  accN += (long long)adc_combine(dnB, upper) << sh;

  accP += shflxor64(accP, 1);  accN += shflxor64(accN, 1);
  accP += shflxor64(accP, 2);  accN += shflxor64(accN, 2);
  accP += shflxor64(accP, 4);  accN += shflxor64(accN, 4);
  accP += shflxor64(accP, 16); accN += shflxor64(accN, 16);

  const double scale = (192.0 / 255.0) / 16777216.0;
  const float op = (float)((double)accP * scale);
  const float on = (float)((double)accN * scale);
  const float tpq = acm_q(op);
  const float tnq = acm_q(on);
  const int o = ntile * 2 + ((lane >> 3) & 1);
  if (lane == 0 || lane == 8)
    out[(size_t)b * OUT_F + o] = (tpq - tnq) * (1.0f / 4096.0f) + bias[o];
}

extern "C" void kernel_launch(void* const* d_in, const int* in_sizes, int n_in,
                              void* d_out, int out_size, void* d_ws, size_t ws_size,
                              hipStream_t stream) {
  const float* x    = (const float*)d_in[0];
  const float* w    = (const float*)d_in[1];
  const float* bias = (const float*)d_in[2];
  float* out = (float*)d_out;

  uint32_t* apk = (uint32_t*)d_ws;
  uint32_t* bpk = apk + (size_t)B_SZ * XR * 256;

  prep_bits  <<<B_SZ * XR,    256, 0, stream>>>(x, apk);
  prep_slices<<<XR * NTILES,  256, 0, stream>>>(w, bpk);
  mvm_wmma   <<<NTILES * 4,   256, 0, stream>>>(apk, bpk, bias, out);
}
